// MTCCtrModel_38920993636411
// MI455X (gfx1250) — matrix operand order repaired and checked by reading, not compiled, not run on hardware
//
#include <hip/hip_runtime.h>

// ---------------------------------------------------------------------------
// MTCC CTR model forward for MI455X (gfx1250).
// All large GEMMs run on v_wmma_f32_16x16x32_bf16 (f32 accumulate). Operands
// are kept in bf16: weights are pre-converted once per launch, activations are
// emitted in bf16 by their producers (BN / mix / attention / GEMM epilogue),
// so the GEMM inner loop is pure global_load_b128 -> v_wmma (no VALU packing).
// Each wave owns a 32x16 C block (two tiles sharing one B fragment).
// Compute-bound workload (~115 GFLOP); working set sits in the 192MB L2, an
// HBM sweep at 23.3 TB/s is ~11us -> matrix-core bound, so minimize non-WMMA
// instructions per WMMA.
// ---------------------------------------------------------------------------

#define BATCH 4096

typedef __attribute__((ext_vector_type(16))) __bf16 v16bf;
typedef __attribute__((ext_vector_type(8)))  float  v8f;

__device__ __forceinline__ unsigned bfbits(float f) {
  union { float f; unsigned u; } x; x.f = f;
  return (x.u + 0x7FFFu + ((x.u >> 16) & 1u)) >> 16;   // round-nearest-even
}
__device__ __forceinline__ unsigned pack2(float lo, float hi) {
  return bfbits(lo) | (bfbits(hi) << 16);
}
__device__ __forceinline__ __bf16 tobf(float f) {
  union { unsigned short s; __bf16 b; } y; y.s = (unsigned short)bfbits(f);
  return y.b;
}

// f32 -> packed bf16 pairs (RNE); n2 = number of pairs
__global__ void mtcc_cvt_bf16(const float* __restrict__ src, unsigned* __restrict__ dst, int n2)
{
  const int t = blockIdx.x * blockDim.x + threadIdx.x;
  if (t >= n2) return;
  const float2 v = *(const float2*)(src + 2 * (size_t)t);
  dst[t] = pack2(v.x, v.y);
}

// ---------------------------------------------------------------------------
// Batched GEMM:  C[e] = A[e] (MxK bf16, row stride lda) * W[e]^T (NxK bf16)
//                + bias;  optional bf16 copy of C for GEMM->GEMM chains.
// grid = (ceil(N/64), M/32, E), block = 128 (4 waves, each a 32x16 C block).
// Requires K%32==0, N%16==0, M%32==0 (true for all uses).
// ---------------------------------------------------------------------------
__global__ __launch_bounds__(128)
void mtcc_wmma_gemm(const __bf16* __restrict__ A, int lda, long long aStride,
                    const __bf16* __restrict__ W, long long wStride,
                    const float* __restrict__ bias, int biasStride,
                    float* __restrict__ C, __bf16* __restrict__ C16,
                    int ldc, long long cStride, int N, int K)
{
  const int e = blockIdx.z;
  A += (size_t)e * aStride;
  W += (size_t)e * wStride;
  C += (size_t)e * cStride;
  if (C16) C16 += (size_t)e * cStride;
  const float* bp = bias ? bias + (size_t)e * biasStride : nullptr;

  const int wave = threadIdx.x >> 5;
  const int lane = threadIdx.x & 31;
  const int tileN = (blockIdx.x * 4 + wave) * 16;
  const int tileM = blockIdx.y * 32;         // two 16-row tiles per wave
  if (tileN >= N) return;                    // wave-uniform; EXEC stays all-ones

  const int hv  = lane >> 4;                 // wave half: 0 or 1
  const int l15 = lane & 15;

  const __bf16* arow0 = A + (size_t)(tileM + l15) * lda;   // rows m..m+15
  const __bf16* arow1 = arow0 + (size_t)16 * lda;          // rows m+16..m+31
  const __bf16* wrow  = W + (size_t)(tileN + l15) * K;     // B column
  const int akb = hv * 8;                    // A: lanes 0-15 -> K{0..7,16..23}, 16-31 -> K{8..15,24..31}
  const int bkb = hv * 8;                    // B: lanes 0-15 -> K{0..7,16..23}, 16-31 -> K{8..15,24..31}

  union Frag { v16bf v; uint4 q[2]; };
  v8f acc0 = {}, acc1 = {};
  for (int k0 = 0; k0 < K; k0 += 32) {
    __builtin_prefetch(wrow + k0 + 256, 0, 0);
    __builtin_prefetch(arow0 + k0 + 256, 0, 0);
    Frag bfr, af0, af1;
    bfr.q[0] = *(const uint4*)(wrow + k0 + bkb);          // 8 bf16, K ascending
    bfr.q[1] = *(const uint4*)(wrow + k0 + 16 + bkb);
    af0.q[0] = *(const uint4*)(arow0 + k0 + akb);
    af0.q[1] = *(const uint4*)(arow0 + k0 + 16 + akb);
    af1.q[0] = *(const uint4*)(arow1 + k0 + akb);
    af1.q[1] = *(const uint4*)(arow1 + k0 + 16 + akb);
    acc0 = __builtin_amdgcn_wmma_f32_16x16x32_bf16(false, af0.v, false, bfr.v,
                                                   (short)0, acc0, false, false);
    acc1 = __builtin_amdgcn_wmma_f32_16x16x32_bf16(false, af1.v, false, bfr.v,
                                                   (short)0, acc1, false, false);
  }

  const int ccol = tileN + l15;
  const float bv = bp ? bp[ccol] : 0.0f;
#pragma unroll
  for (int v = 0; v < 8; ++v) {
    // C layout: VGPR v -> M = v (lanes 0-15) / 8+v (lanes 16-31); N = lane&15
    const int row = tileM + hv * 8 + v;
    const float c0 = acc0[v] + bv;
    const float c1 = acc1[v] + bv;
    C[(size_t)row * ldc + ccol]        = c0;
    C[(size_t)(row + 16) * ldc + ccol] = c1;
    if (C16) {
      C16[(size_t)row * ldc + ccol]        = tobf(c0);
      C16[(size_t)(row + 16) * ldc + ccol] = tobf(c1);
    }
  }
}

// ---------------------------------------------------------------------------
// BatchNorm1d (training-mode batch stats) + ReLU, in place; optional bf16 copy
// of the output when it feeds a WMMA GEMM. One block per column.
// ---------------------------------------------------------------------------
__global__ void mtcc_bn_relu(float* __restrict__ h, __bf16* __restrict__ o16,
                             const float* __restrict__ g, const float* __restrict__ bt,
                             int M, int C)
{
  const int c = blockIdx.x;
  __shared__ float s1[256], s2[256];
  float sum = 0.f, sq = 0.f;
  for (int r = threadIdx.x; r < M; r += blockDim.x) {
    float v = h[(size_t)r * C + c];
    sum += v; sq += v * v;
  }
  s1[threadIdx.x] = sum; s2[threadIdx.x] = sq;
  __syncthreads();
  for (int s = 128; s > 0; s >>= 1) {
    if ((int)threadIdx.x < s) { s1[threadIdx.x] += s1[threadIdx.x + s]; s2[threadIdx.x] += s2[threadIdx.x + s]; }
    __syncthreads();
  }
  const float mean = s1[0] / (float)M;
  const float var  = s2[0] / (float)M - mean * mean;
  const float inv  = rsqrtf(var + 1e-5f);
  const float ga = g[c], be = bt[c];
  for (int r = threadIdx.x; r < M; r += blockDim.x) {
    float v = (h[(size_t)r * C + c] - mean) * inv * ga + be;
    v = v > 0.f ? v : 0.f;
    h[(size_t)r * C + c] = v;
    if (o16) o16[(size_t)r * C + c] = tobf(v);
  }
}

// ---------------------------------------------------------------------------
// Small dense linear: C(MxN) = A(MxK) * W(NxK)^T + bias. Thread per (m,n).
// ---------------------------------------------------------------------------
__global__ void mtcc_linsmall(const float* __restrict__ A, int lda,
                              const float* __restrict__ W, const float* __restrict__ bias,
                              float* __restrict__ C, int M, int N, int K)
{
  const int t = blockIdx.x * blockDim.x + threadIdx.x;
  if (t >= M * N) return;
  const int m = t / N, n = t % N;
  const float* a = A + (size_t)m * lda;
  const float* w = W + (size_t)n * K;
  float acc = bias ? bias[n] : 0.f;
  for (int k = 0; k < K; ++k) acc += a[k] * w[k];
  C[t] = acc;
}

__global__ void mtcc_softmax(float* __restrict__ g, int M, int E)
{
  const int m = blockIdx.x * blockDim.x + threadIdx.x;
  if (m >= M) return;
  float* row = g + (size_t)m * E;
  float mx = row[0];
  for (int e = 1; e < E; ++e) mx = fmaxf(mx, row[e]);
  float s = 0.f;
  for (int e = 0; e < E; ++e) { float v = __expf(row[e] - mx); row[e] = v; s += v; }
  const float inv = 1.f / s;
  for (int e = 0; e < E; ++e) row[e] *= inv;
}

// out(b,d) = sum_e gate(b,e) * concat(src0,src1,src2)(b,e,d); optional bf16 copy
__global__ void mtcc_mix(const float* __restrict__ gate, int Etot,
                         const float* __restrict__ s0, int e0,
                         const float* __restrict__ s1, int e1,
                         const float* __restrict__ s2, int e2,
                         int D, float* __restrict__ out, __bf16* __restrict__ o16)
{
  const int t = blockIdx.x * blockDim.x + threadIdx.x;
  if (t >= BATCH * D) return;
  const int b = t / D, d = t % D;
  const float* grow = gate + (size_t)b * Etot;
  float acc = 0.f; int gi = 0;
  for (int e = 0; e < e0; ++e) acc += grow[gi++] * s0[((size_t)b * e0 + e) * D + d];
  if (s1) for (int e = 0; e < e1; ++e) acc += grow[gi++] * s1[((size_t)b * e1 + e) * D + d];
  if (s2) for (int e = 0; e < e2; ++e) acc += grow[gi++] * s2[((size_t)b * e2 + e) * D + d];
  out[t] = acc;
  if (o16) o16[t] = tobf(acc);
}

// ---------------------------------------------------------------------------
// Gather kernels
// ---------------------------------------------------------------------------
__global__ void mtcc_embed(const int* __restrict__ x, const float* __restrict__ embed,
                           float* __restrict__ feat, __bf16* __restrict__ feat16)
{
  const int t = blockIdx.x * blockDim.x + threadIdx.x;
  if (t >= BATCH * 1024) return;
  const int b = t >> 10, r = t & 1023, f = r >> 5, d = r & 31;
  const int idx = x[b * 32 + f] + f * 1000;
  const float v = embed[(size_t)idx * 32 + d];
  feat[t] = v;
  feat16[t] = tobf(v);
}

__global__ void mtcc_linx(const int* __restrict__ x, const float* __restrict__ lw,
                          const float* __restrict__ lb, float* __restrict__ linx)
{
  const int b = blockIdx.x * blockDim.x + threadIdx.x;
  if (b >= BATCH) return;
  float s = lb[0];
  for (int f = 0; f < 32; ++f) s += lw[x[b * 32 + f] + f * 1000];
  linx[b] = s;
}

// ---------------------------------------------------------------------------
// AutoInt attention: one block per (b, head); F=32, head_dim=32, 2 heads.
// qkv layout (B*32, 192) = [q(64) | k(64) | v(64)]; output written in bf16
// (it only feeds the next WMMA GEMM).
// ---------------------------------------------------------------------------
__global__ void mtcc_attention(const float* __restrict__ qkv, __bf16* __restrict__ attn16)
{
  const int b = blockIdx.x >> 1;
  const int h = blockIdx.x & 1;
  __shared__ float qs[32][33], ks[32][33], vs[32][33], ss[32][33];
  const float* base = qkv + (size_t)b * 32 * 192;
  for (int i = threadIdx.x; i < 32 * 32; i += blockDim.x) {
    const int f = i >> 5, d = i & 31;
    qs[f][d] = base[f * 192 +       h * 32 + d];
    ks[f][d] = base[f * 192 +  64 + h * 32 + d];
    vs[f][d] = base[f * 192 + 128 + h * 32 + d];
  }
  __syncthreads();
  const float scale = 0.17677669529663687f;   // 32^-0.5
  for (int i = threadIdx.x; i < 32 * 32; i += blockDim.x) {
    const int f = i >> 5, gcol = i & 31;
    float acc = 0.f;
    for (int d = 0; d < 32; ++d) acc += qs[f][d] * ks[gcol][d];
    ss[f][gcol] = acc * scale;
  }
  __syncthreads();
  for (int f = threadIdx.x; f < 32; f += blockDim.x) {
    float mx = ss[f][0];
    for (int gcol = 1; gcol < 32; ++gcol) mx = fmaxf(mx, ss[f][gcol]);
    float s = 0.f;
    for (int gcol = 0; gcol < 32; ++gcol) { float v = __expf(ss[f][gcol] - mx); ss[f][gcol] = v; s += v; }
    const float inv = 1.f / s;
    for (int gcol = 0; gcol < 32; ++gcol) ss[f][gcol] *= inv;
  }
  __syncthreads();
  __bf16* ob = attn16 + (size_t)b * 32 * 64;
  for (int i = threadIdx.x; i < 32 * 32; i += blockDim.x) {
    const int f = i >> 5, d = i & 31;
    float acc = 0.f;
    for (int gcol = 0; gcol < 32; ++gcol) acc += ss[f][gcol] * vs[gcol][d];
    ob[f * 64 + h * 32 + d] = tobf(acc);
  }
}

__global__ void mtcc_addrelu(float* __restrict__ a, const float* __restrict__ v, int n)
{
  const int t = blockIdx.x * blockDim.x + threadIdx.x;
  if (t >= n) return;
  a[t] = fmaxf(a[t] + v[t], 0.f);
}

// att_x(b) = cross(b,:) . fc_w + fc_b  (2048-wide reduction per row)
__global__ void mtcc_attx(const float* __restrict__ cross, const float* __restrict__ fcw,
                          const float* __restrict__ fcb, float* __restrict__ attx)
{
  const int b = blockIdx.x;
  __shared__ float red[256];
  const float* row = cross + (size_t)b * 2048;
  float acc = 0.f;
  for (int i = threadIdx.x; i < 2048; i += blockDim.x) acc += row[i] * fcw[i];
  red[threadIdx.x] = acc;
  __syncthreads();
  for (int s = 128; s > 0; s >>= 1) {
    if ((int)threadIdx.x < s) red[threadIdx.x] += red[threadIdx.x + s];
    __syncthreads();
  }
  if (threadIdx.x == 0) attx[b] = red[0] + fcb[0];
}

// guide(b) = sigmoid(tf0(b,:) . w + b)
__global__ void mtcc_guide(const float* __restrict__ tf0, const float* __restrict__ gw,
                           const float* __restrict__ gb, float* __restrict__ g)
{
  const int b = blockIdx.x * blockDim.x + threadIdx.x;
  if (b >= BATCH) return;
  float s = gb[0];
  const float* row = tf0 + (size_t)b * 128;
  for (int i = 0; i < 128; ++i) s += gw[i] * row[i];
  g[b] = 1.f / (1.f + __expf(-s));
}

// unin(b,:) = [ tf2L1(b, 0:128) | guide(b) * tower0(b, 0:64) ]  (+ bf16 copy)
__global__ void mtcc_ctr_concat(const float* __restrict__ g, const float* __restrict__ t0,
                                const float* __restrict__ tf2,
                                float* __restrict__ unin, __bf16* __restrict__ unin16)
{
  const int t = blockIdx.x * blockDim.x + threadIdx.x;
  if (t >= BATCH * 192) return;
  const int b = t / 192, i = t % 192;
  const float v = (i < 128) ? tf2[(size_t)b * 128 + i]
                            : g[b] * t0[(size_t)b * 64 + (i - 128)];
  unin[t] = v;
  unin16[t] = tobf(v);
}

// ---------------------------------------------------------------------------
// Output heads: res0 (B,), res1 (B,9), final = sg(res1) - united_res (B,9)
// ---------------------------------------------------------------------------
__global__ void mtcc_heads(const float* __restrict__ linx, const float* __restrict__ attx,
                           const float* __restrict__ tw0, const float* __restrict__ tw1,
                           const float* __restrict__ uni,
                           const float* __restrict__ w0a, const float* __restrict__ b0a,
                           const float* __restrict__ w0b, const float* __restrict__ b0b,
                           const float* __restrict__ w1a, const float* __restrict__ b1a,
                           const float* __restrict__ w1b, const float* __restrict__ b1b,
                           const float* __restrict__ wua, const float* __restrict__ bua,
                           const float* __restrict__ wub, const float* __restrict__ bub,
                           float* __restrict__ out)
{
  const int b = blockIdx.x * blockDim.x + threadIdx.x;
  if (b >= BATCH) return;
  const float xa = linx[b], xb = attx[b];
  // --- out0: (66 -> 8 -> 1) sigmoid ---
  float h0[8];
  for (int j = 0; j < 8; ++j) {
    const float* w = w0a + j * 66;
    float a = b0a[j] + w[0] * xa + w[1] * xb;
    for (int i = 0; i < 64; ++i) a += w[2 + i] * tw0[(size_t)b * 64 + i];
    h0[j] = a;
  }
  float z = b0b[0];
  for (int j = 0; j < 8; ++j) z += w0b[j] * h0[j];
  out[b] = 1.f / (1.f + __expf(-z));
  // --- out1: (66 -> 9 -> 9) ---
  float h1[9], r1[9];
  for (int j = 0; j < 9; ++j) {
    const float* w = w1a + j * 66;
    float a = b1a[j] + w[0] * xa + w[1] * xb;
    for (int i = 0; i < 64; ++i) a += w[2 + i] * tw1[(size_t)b * 64 + i];
    h1[j] = a;
  }
  for (int t = 0; t < 9; ++t) {
    float a = b1b[t];
    for (int j = 0; j < 9; ++j) a += w1b[t * 9 + j] * h1[j];
    r1[t] = a;
    out[BATCH + (size_t)b * 9 + t] = a;
  }
  // --- out_u on [lin, att, united] ---
  float hu[9];
  for (int j = 0; j < 9; ++j) {
    const float* w = wua + j * 66;
    float a = bua[j] + w[0] * xa + w[1] * xb;
    for (int i = 0; i < 64; ++i) a += w[2 + i] * uni[(size_t)b * 64 + i];
    hu[j] = a;
  }
  for (int t = 0; t < 9; ++t) {
    float a = bub[t];
    for (int j = 0; j < 9; ++j) a += wub[t * 9 + j] * hu[j];
    out[BATCH * 10 + (size_t)b * 9 + t] = r1[t] - a;   // final = sg(res1) - united_res
  }
}

// ---------------------------------------------------------------------------
// Host-side launch. Input order = jax pytree leaves (sorted dict keys).
// ---------------------------------------------------------------------------
extern "C" void kernel_launch(void* const* d_in, const int* in_sizes, int n_in,
                              void* d_out, int out_size, void* d_ws, size_t ws_size,
                              hipStream_t stream)
{
  (void)in_sizes; (void)n_in; (void)out_size; (void)ws_size;
  int cur = 0;
  auto nf = [&]() -> const float* { return (const float*)d_in[cur++]; };

  // params.atten { ae{b,w}, fc{b,w}, layers[3]{inp{b,w},out{b,w}}, vres{b,w} }
  const float* ae_b = nf();  const float* ae_w = nf();
  const float* fc_b = nf();  const float* fc_w = nf();
  const float *inp_b[3], *inp_w[3], *aout_b[3], *aout_w[3];
  for (int l = 0; l < 3; ++l) { inp_b[l] = nf(); inp_w[l] = nf(); aout_b[l] = nf(); aout_w[l] = nf(); }
  const float* vres_b = nf(); const float* vres_w = nf();
  // params.embed
  const float* embed = nf();
  // params.guide {b,w}
  const float* guide_b = nf(); const float* guide_w = nf();
  // params.lin_b, params.lin_w
  const float* lin_b = nf(); const float* lin_w = nf();
  // params.out0 [lin(8,66), lin(1,8)]
  const float* o0a_b = nf(); const float* o0a_w = nf();
  const float* o0b_b = nf(); const float* o0b_w = nf();
  // params.out1 [lin(9,66), lin(9,9)]
  const float* o1a_b = nf(); const float* o1a_w = nf();
  const float* o1b_b = nf(); const float* o1b_w = nf();
  // params.out_u
  const float* oua_b = nf(); const float* oua_w = nf();
  const float* oub_b = nf(); const float* oub_w = nf();
  // params.shared_experts [layer][sublayer]{b,bt,g,w}
  const float *se_b[2][2], *se_bt[2][2], *se_g[2][2], *se_w[2][2];
  for (int i = 0; i < 2; ++i) for (int s = 0; s < 2; ++s) {
    se_b[i][s] = nf(); se_bt[i][s] = nf(); se_g[i][s] = nf(); se_w[i][s] = nf();
  }
  // params.shared_gates [lin(16,1024), lin(8,256)]
  const float *sg_b[2], *sg_w[2];
  for (int i = 0; i < 2; ++i) { sg_b[i] = nf(); sg_w[i] = nf(); }
  // params.task_experts [layer][task][sublayer]{b,bt,g,w}
  const float *te_b[2][2][2], *te_bt[2][2][2], *te_g[2][2][2], *te_w[2][2][2];
  for (int i = 0; i < 2; ++i) for (int j = 0; j < 2; ++j) for (int s = 0; s < 2; ++s) {
    te_b[i][j][s] = nf(); te_bt[i][j][s] = nf(); te_g[i][j][s] = nf(); te_w[i][j][s] = nf();
  }
  // params.task_gates [layer][task]{b,w}
  const float *tg_b[2][2], *tg_w[2][2];
  for (int i = 0; i < 2; ++i) for (int j = 0; j < 2; ++j) { tg_b[i][j] = nf(); tg_w[i][j] = nf(); }
  // params.towers [task][sublayer]{b,bt,g,w}
  const float *tw_b[2][2], *tw_bt[2][2], *tw_g[2][2], *tw_w[2][2];
  for (int t = 0; t < 2; ++t) for (int s = 0; s < 2; ++s) {
    tw_b[t][s] = nf(); tw_bt[t][s] = nf(); tw_g[t][s] = nf(); tw_w[t][s] = nf();
  }
  // params.united_tower [sublayer]{b,bt,g,w}
  const float *ut_b[2], *ut_bt[2], *ut_g[2], *ut_w[2];
  for (int s = 0; s < 2; ++s) { ut_b[s] = nf(); ut_bt[s] = nf(); ut_g[s] = nf(); ut_w[s] = nf(); }
  // x
  const int* x = (const int*)d_in[cur++];

  // -------------------- workspace (bump allocation, float units) ------------
  float* wsf = (float*)d_ws;
  size_t off = 0;
  auto take = [&](size_t n) { float* p = wsf + off; off += n; return p; };
  auto takebf = [&](size_t nbf) { float* p = wsf + off; off += nbf / 2; return (__bf16*)p; };
  const size_t Bz = BATCH;

  // f32 buffers
  float* FEAT   = take(Bz * 1024);
  float* LINX   = take(Bz);
  float* ATTX   = take(Bz);
  float* TF0L0  = take(Bz * 256);
  float* TF1L0  = take(Bz * 256);
  float* TF2L0  = take(Bz * 256);
  float* TF0L1  = take(Bz * 128);
  float* TF1L1  = take(Bz * 128);
  float* TF2L1  = take(Bz * 128);
  float* GATE   = take(Bz * 16);
  float* TW0H   = take(Bz * 128);
  float* TW0    = take(Bz * 64);
  float* TW1H   = take(Bz * 128);
  float* TW1    = take(Bz * 64);
  float* GUIDE  = take(Bz);
  float* UNIN   = take(Bz * 192);
  float* UNH    = take(Bz * 128);
  float* UNITED = take(Bz * 64);

  // bf16 activation buffers
  __bf16* FEAT16   = takebf(Bz * 1024);
  __bf16* TF0L0b   = takebf(Bz * 256);
  __bf16* TF1L0b   = takebf(Bz * 256);
  __bf16* TF2L0b   = takebf(Bz * 256);
  __bf16* TF0L1b   = takebf(Bz * 128);
  __bf16* TF1L1b   = takebf(Bz * 128);
  __bf16* TW0Hb    = takebf(Bz * 128);
  __bf16* TW1Hb    = takebf(Bz * 128);
  __bf16* UNINb    = takebf(Bz * 192);
  __bf16* UNHb     = takebf(Bz * 128);

  // bf16 weight copies
  auto cvt = [&](const float* src, size_t n) -> const __bf16* {
    __bf16* dst = takebf(n);
    mtcc_cvt_bf16<<<((unsigned)(n / 2) + 255) / 256, 256, 0, stream>>>(src, (unsigned*)dst, (int)(n / 2));
    return dst;
  };
  const __bf16 *se_wb[2][2], *te_wb[2][2][2], *tw_wb[2][2], *ut_wb[2];
  se_wb[0][0] = cvt(se_w[0][0], 8ull * 512 * 1024);
  se_wb[0][1] = cvt(se_w[0][1], 8ull * 256 * 512);
  se_wb[1][0] = cvt(se_w[1][0], 8ull * 256 * 256);
  se_wb[1][1] = cvt(se_w[1][1], 8ull * 128 * 256);
  for (int t = 0; t < 2; ++t) {
    te_wb[0][t][0] = cvt(te_w[0][t][0], 4ull * 512 * 1024);
    te_wb[0][t][1] = cvt(te_w[0][t][1], 4ull * 256 * 512);
    te_wb[1][t][0] = cvt(te_w[1][t][0], 4ull * 256 * 256);
    te_wb[1][t][1] = cvt(te_w[1][t][1], 4ull * 128 * 256);
    tw_wb[t][0] = cvt(tw_w[t][0], 128ull * 128);
    tw_wb[t][1] = cvt(tw_w[t][1], 64ull * 128);
  }
  ut_wb[0] = cvt(ut_w[0], 128ull * 192);
  ut_wb[1] = cvt(ut_w[1], 64ull * 128);
  const __bf16* ae_wb   = cvt(ae_w, 64ull * 32);
  const __bf16* vres_wb = cvt(vres_w, 64ull * 32);
  const __bf16 *inp_wb[3], *aout_wb[3];
  for (int l = 0; l < 3; ++l) {
    inp_wb[l]  = cvt(inp_w[l], 192ull * 64);
    aout_wb[l] = cvt(aout_w[l], 64ull * 64);
  }

  // big reusable region: expert phase f32 + bf16, then autoint phase
  float* BIG = wsf + off;
  // expert-phase f32 views
  float* SH_H = BIG;
  float* SH_O = SH_H + Bz * 8 * 512;
  float* T0H  = SH_O + Bz * 8 * 256;
  float* T0O  = T0H  + Bz * 4 * 512;
  float* T1H  = T0O  + Bz * 4 * 256;
  float* T1O  = T1H  + Bz * 4 * 512;
  float* BIG2 = T1O  + Bz * 4 * 256;          // second region (bf16 hidden copies)
  __bf16* SH_Hb = (__bf16*)BIG2;              // B*4096 bf16
  __bf16* T0Hb  = SH_Hb + Bz * 8 * 512;       // B*2048 bf16
  __bf16* T1Hb  = T0Hb + Bz * 4 * 512;
  // autoint-phase views of BIG (experts are dead by then)
  float* QKV  = BIG;
  float* ABUF = QKV  + Bz * 32 * 192;
  float* A2   = ABUF + Bz * 32 * 64;
  float* VRES = A2   + Bz * 32 * 64;
  __bf16* ABUFb = (__bf16*)(VRES + Bz * 32 * 64);
  __bf16* A2b   = ABUFb + Bz * 32 * 64;
  __bf16* ATTNb = A2b + Bz * 32 * 64;

  // -------------------- launch helpers --------------------------------------
  auto gemm = [&](const __bf16* A, int lda, long long as, const __bf16* W, long long wst,
                  const float* bias, int bst, float* Cp, __bf16* C16, int ldc, long long cst,
                  int M, int N, int K, int E) {
    dim3 grid((N + 63) / 64, M / 32, E);
    mtcc_wmma_gemm<<<grid, 128, 0, stream>>>(A, lda, as, W, wst, bias, bst, Cp, C16, ldc, cst, N, K);
  };
  auto bn = [&](float* h, __bf16* o16, const float* g, const float* bt, int C) {
    mtcc_bn_relu<<<C, 256, 0, stream>>>(h, o16, g, bt, BATCH, C);
  };
  auto lins = [&](const float* A, int lda, const float* W, const float* bias, float* Cp, int N, int K) {
    mtcc_linsmall<<<(BATCH * N + 255) / 256, 256, 0, stream>>>(A, lda, W, bias, Cp, BATCH, N, K);
  };
  auto smax = [&](float* g, int E) {
    mtcc_softmax<<<(BATCH + 255) / 256, 256, 0, stream>>>(g, BATCH, E);
  };
  auto mix = [&](const float* g, int Et, const float* s0, int e0, const float* s1, int e1,
                 const float* s2, int e2, int D, float* outp, __bf16* o16) {
    mtcc_mix<<<(BATCH * D + 255) / 256, 256, 0, stream>>>(g, Et, s0, e0, s1, e1, s2, e2, D, outp, o16);
  };

  // -------------------- phase 0: embeddings + linear term -------------------
  mtcc_embed<<<(BATCH * 1024 + 255) / 256, 256, 0, stream>>>(x, embed, FEAT, FEAT16);
  mtcc_linx<<<(BATCH + 255) / 256, 256, 0, stream>>>(x, lin_w, lin_b, LINX);

  // -------------------- PLE layer 0 (in=1024, dims 512->256) ----------------
  gemm(FEAT16, 1024, 0, se_wb[0][0], 512LL * 1024, se_b[0][0], 512, SH_H, nullptr, 4096, 512, BATCH, 512, 1024, 8);
  bn(SH_H, SH_Hb, se_g[0][0], se_bt[0][0], 4096);
  gemm(SH_Hb, 4096, 512, se_wb[0][1], 256LL * 512, se_b[0][1], 256, SH_O, nullptr, 2048, 256, BATCH, 256, 512, 8);
  bn(SH_O, nullptr, se_g[0][1], se_bt[0][1], 2048);
  for (int t = 0; t < 2; ++t) {
    float* TH = t ? T1H : T0H;   __bf16* THb = t ? T1Hb : T0Hb;
    float* TO = t ? T1O : T0O;
    float* TF = t ? TF1L0 : TF0L0;   __bf16* TFb = t ? TF1L0b : TF0L0b;
    gemm(FEAT16, 1024, 0, te_wb[0][t][0], 512LL * 1024, te_b[0][t][0], 512, TH, nullptr, 2048, 512, BATCH, 512, 1024, 4);
    bn(TH, THb, te_g[0][t][0], te_bt[0][t][0], 2048);
    gemm(THb, 2048, 512, te_wb[0][t][1], 256LL * 512, te_b[0][t][1], 256, TO, nullptr, 1024, 256, BATCH, 256, 512, 4);
    bn(TO, nullptr, te_g[0][t][1], te_bt[0][t][1], 1024);
    lins(FEAT, 1024, tg_w[0][t], tg_b[0][t], GATE, 12, 1024);
    smax(GATE, 12);
    mix(GATE, 12, TO, 4, SH_O, 8, nullptr, 0, 256, TF, TFb);
  }
  lins(FEAT, 1024, sg_w[0], sg_b[0], GATE, 16, 1024);
  smax(GATE, 16);
  mix(GATE, 16, T0O, 4, T1O, 4, SH_O, 8, 256, TF2L0, TF2L0b);

  // -------------------- PLE layer 1 (in=256, dims 256->128) -----------------
  gemm(TF2L0b, 256, 0, se_wb[1][0], 256LL * 256, se_b[1][0], 256, SH_H, nullptr, 2048, 256, BATCH, 256, 256, 8);
  bn(SH_H, SH_Hb, se_g[1][0], se_bt[1][0], 2048);
  gemm(SH_Hb, 2048, 256, se_wb[1][1], 128LL * 256, se_b[1][1], 128, SH_O, nullptr, 1024, 128, BATCH, 128, 256, 8);
  bn(SH_O, nullptr, se_g[1][1], se_bt[1][1], 1024);
  for (int t = 0; t < 2; ++t) {
    float* TH = t ? T1H : T0H;   __bf16* THb = t ? T1Hb : T0Hb;
    float* TO = t ? T1O : T0O;
    const __bf16* TFinb = t ? TF1L0b : TF0L0b;
    const float* TFin = t ? TF1L0 : TF0L0;
    float* TFout = t ? TF1L1 : TF0L1;   __bf16* TFoutb = t ? TF1L1b : TF0L1b;
    gemm(TFinb, 256, 0, te_wb[1][t][0], 256LL * 256, te_b[1][t][0], 256, TH, nullptr, 1024, 256, BATCH, 256, 256, 4);
    bn(TH, THb, te_g[1][t][0], te_bt[1][t][0], 1024);
    gemm(THb, 1024, 256, te_wb[1][t][1], 128LL * 256, te_b[1][t][1], 128, TO, nullptr, 512, 128, BATCH, 128, 256, 4);
    bn(TO, nullptr, te_g[1][t][1], te_bt[1][t][1], 512);
    lins(TFin, 256, tg_w[1][t], tg_b[1][t], GATE, 12, 256);
    smax(GATE, 12);
    mix(GATE, 12, TO, 4, SH_O, 8, nullptr, 0, 128, TFout, TFoutb);
  }
  lins(TF2L0, 256, sg_w[1], sg_b[1], GATE, 8, 256);
  smax(GATE, 8);
  mix(GATE, 8, SH_O, 8, nullptr, 0, nullptr, 0, 128, TF2L1, nullptr);  // mix_s = sg(shared_out)

  // -------------------- towers + guide + united -----------------------------
  gemm(TF0L1b, 128, 0, tw_wb[0][0], 0, tw_b[0][0], 0, TW0H, nullptr, 128, 0, BATCH, 128, 128, 1);
  bn(TW0H, TW0Hb, tw_g[0][0], tw_bt[0][0], 128);
  gemm(TW0Hb, 128, 0, tw_wb[0][1], 0, tw_b[0][1], 0, TW0, nullptr, 64, 0, BATCH, 64, 128, 1);
  bn(TW0, nullptr, tw_g[0][1], tw_bt[0][1], 64);
  gemm(TF1L1b, 128, 0, tw_wb[1][0], 0, tw_b[1][0], 0, TW1H, nullptr, 128, 0, BATCH, 128, 128, 1);
  bn(TW1H, TW1Hb, tw_g[1][0], tw_bt[1][0], 128);
  gemm(TW1Hb, 128, 0, tw_wb[1][1], 0, tw_b[1][1], 0, TW1, nullptr, 64, 0, BATCH, 64, 128, 1);
  bn(TW1, nullptr, tw_g[1][1], tw_bt[1][1], 64);

  mtcc_guide<<<(BATCH + 255) / 256, 256, 0, stream>>>(TF0L1, guide_w, guide_b, GUIDE);
  mtcc_ctr_concat<<<(BATCH * 192 + 255) / 256, 256, 0, stream>>>(GUIDE, TW0, TF2L1, UNIN, UNINb);
  gemm(UNINb, 192, 0, ut_wb[0], 0, ut_b[0], 0, UNH, nullptr, 128, 0, BATCH, 128, 192, 1);
  bn(UNH, UNHb, ut_g[0], ut_bt[0], 128);
  gemm(UNHb, 128, 0, ut_wb[1], 0, ut_b[1], 0, UNITED, nullptr, 64, 0, BATCH, 64, 128, 1);
  bn(UNITED, nullptr, ut_g[1], ut_bt[1], 64);

  // -------------------- AutoInt ---------------------------------------------
  const int M32 = BATCH * 32;
  gemm(FEAT16, 32, 0, ae_wb, 0, ae_b, 0, ABUF, ABUFb, 64, 0, M32, 64, 32, 1);
  float* acur = ABUF;  __bf16* acurb = ABUFb;
  float* anxt = A2;    __bf16* anxtb = A2b;
  for (int l = 0; l < 3; ++l) {
    gemm(acurb, 64, 0, inp_wb[l], 0, inp_b[l], 0, QKV, nullptr, 192, 0, M32, 192, 64, 1);
    mtcc_attention<<<BATCH * 2, 128, 0, stream>>>(QKV, ATTNb);
    gemm(ATTNb, 64, 0, aout_wb[l], 0, aout_b[l], 0, anxt, anxtb, 64, 0, M32, 64, 64, 1);
    float* tf = acur; acur = anxt; anxt = tf;
    __bf16* tb = acurb; acurb = anxtb; anxtb = tb;
  }
  gemm(FEAT16, 32, 0, vres_wb, 0, vres_b, 0, VRES, nullptr, 64, 0, M32, 64, 32, 1);
  mtcc_addrelu<<<(M32 * 64 + 255) / 256, 256, 0, stream>>>(acur, VRES, M32 * 64);
  mtcc_attx<<<BATCH, 256, 0, stream>>>(acur, fc_w, fc_b, ATTX);

  // -------------------- heads -> d_out --------------------------------------
  mtcc_heads<<<(BATCH + 255) / 256, 256, 0, stream>>>(
      LINX, ATTX, TW0, TW1, UNITED,
      o0a_w, o0a_b, o0b_w, o0b_b,
      o1a_w, o1a_b, o1b_w, o1b_b,
      oua_w, oua_b, oub_w, oub_b,
      (float*)d_out);
}
